// Attention_18631568130256
// MI455X (gfx1250) — compile-verified, matrix operand order checked by reading, not run on hardware
//
#include <hip/hip_runtime.h>

typedef __attribute__((ext_vector_type(16))) __bf16 v16bf;
typedef __attribute__((ext_vector_type(8)))  __bf16 v8bf;
typedef __attribute__((ext_vector_type(4)))  __bf16 v4bf;
typedef __attribute__((ext_vector_type(8)))  float  v8f;
typedef __attribute__((ext_vector_type(4)))  float  v4f;
typedef __attribute__((ext_vector_type(4)))  unsigned v4u;
typedef __attribute__((ext_vector_type(8)))  int    v8i;
typedef __attribute__((ext_vector_type(4)))  int    v4i;

#define WMMA_BF16(a,b,c) \
  __builtin_amdgcn_wmma_f32_16x16x32_bf16(false,(a),false,(b),(short)0,(c),false,false)

#if __has_builtin(__builtin_amdgcn_tensor_load_to_lds)
#define HAVE_TDM 1
#else
#define HAVE_TDM 0
#endif

static __device__ __forceinline__ v16bf combine16(v8bf lo, v8bf hi) {
  return __builtin_shufflevector(lo, hi, 0,1,2,3,4,5,6,7,8,9,10,11,12,13,14,15);
}
static __device__ __forceinline__ v8f v8f_zero() {
  v8f z;
#pragma unroll
  for (int i = 0; i < 8; ++i) z[i] = 0.0f;
  return z;
}

// ---------------------------------------------------------------------------
// CDNA5 async global->LDS copy (ASYNCcnt-tracked, no VGPR staging).
// ---------------------------------------------------------------------------
static __device__ __forceinline__ void async_cp_b128(const void* lds_dst,
                                                     const void* gsrc) {
  unsigned lds_off = (unsigned)(unsigned long long)lds_dst;
  unsigned long long ga = (unsigned long long)gsrc;
  asm volatile("global_load_async_to_lds_b128 %0, %1, off"
               :: "v"(lds_off), "v"(ga) : "memory");
}
static __device__ __forceinline__ void async_wait0() {
  asm volatile("s_wait_asynccnt 0" ::: "memory");
}
static __device__ __forceinline__ void tensor_wait0() {
#if __has_builtin(__builtin_amdgcn_s_wait_tensorcnt)
  __builtin_amdgcn_s_wait_tensorcnt(0);
#else
  asm volatile("s_wait_tensorcnt 0" ::: "memory");
#endif
}

// ---------------------------------------------------------------------------
// TDM: DMA a 2D tile [tile_d1 rows x tile_d0 elems] of bf16 from global into
// LDS, with hardware padding producing the LDS row stride the WMMA fragment
// loads expect. D# packing per CDNA5 ISA §8.3/8.4. Issued by one wave.
//   pad_interval code: dwords-per-burst = 2<<code ; pad_amount code: (code+1) dw
// ---------------------------------------------------------------------------
static __device__ __forceinline__ void tdm_load_2d_bf16(
    unsigned lds_addr, const void* gsrc,
    unsigned tensor_d0, unsigned tensor_d1, unsigned d0_stride,
    unsigned tile_d0, unsigned tile_d1,
    unsigned pad_interval, unsigned pad_amount) {
#if HAVE_TDM
  unsigned long long ga = (unsigned long long)gsrc;
  v4u g0;
  g0[0] = 1u;                                     // count=1, user descriptor
  g0[1] = lds_addr;                               // LDS byte address
  g0[2] = (unsigned)ga;                           // global_addr[31:0]
  g0[3] = (unsigned)((ga >> 32) & 0x1FFFFFFu) | (2u << 30);  // [56:32] | type=2
  v8i g1;
  g1[0] = (int)((1u << 16)                        // data_size = 2 bytes
              | (1u << 20)                        // pad_enable
              | (pad_interval << 22) | (pad_amount << 25));
  g1[1] = (int)((tensor_d0 & 0xFFFFu) << 16);     // tensor_dim0[15:0]
  g1[2] = (int)((tensor_d0 >> 16) | ((tensor_d1 & 0xFFFFu) << 16));
  g1[3] = (int)((tensor_d1 >> 16) | (tile_d0 << 16));
  g1[4] = (int)tile_d1;                           // tile_dim1 (tile_dim2=0)
  g1[5] = (int)d0_stride;                         // tensor_dim0_stride[31:0]
  g1[6] = 0;
  g1[7] = 0;
  v4i zz = {0, 0, 0, 0};
#if __clang_major__ >= 23
  v8i z8 = {0, 0, 0, 0, 0, 0, 0, 0};
  __builtin_amdgcn_tensor_load_to_lds(g0, g1, zz, zz, z8, 0);
#else
  __builtin_amdgcn_tensor_load_to_lds(g0, g1, zz, zz, 0);
#endif
#endif
}

// ---------------------------------------------------------------------------
// fp32 -> bf16 convert (vectorized x4)
// ---------------------------------------------------------------------------
__global__ void __launch_bounds__(256) cvt_f32_bf16(const float* __restrict__ in,
                                                    __bf16* __restrict__ out, int n4) {
  int i = blockIdx.x * blockDim.x + threadIdx.x;
  if (i >= n4) return;
  v4f f = ((const v4f*)in)[i];
  v4bf b;
  b.x = (__bf16)f.x; b.y = (__bf16)f.y; b.z = (__bf16)f.z; b.w = (__bf16)f.w;
  ((v4bf*)out)[i] = b;
}

// ---------------------------------------------------------------------------
// fp32 [K=1024][N] -> bf16 TRANSPOSED [N][1024]
// ---------------------------------------------------------------------------
__global__ void __launch_bounds__(256) cvt_t_f32_bf16(const float* __restrict__ in,
                                                      __bf16* __restrict__ out,
                                                      int N) {
  int o = blockIdx.x * 256 + threadIdx.x;   // linear index into out [N][1024]
  int k = o & 1023;
  int n = o >> 10;
  out[o] = (__bf16)in[(size_t)k * N + n];
}

// ---------------------------------------------------------------------------
// bf16 GEMM: C[M,N] = A[M,1024] * Bt[N,1024]^T + bias
// block tile 128x128, BK=32, 8 waves, wave tile 64x32 (4x2 WMMA accums).
// Tiles async-copied (ASYNCcnt) straight into K-major LDS.
// ---------------------------------------------------------------------------
template <int EPI>
__global__ void __launch_bounds__(256) gemm_bf16(const __bf16* __restrict__ A,
                                                 const __bf16* __restrict__ Bt,
                                                 const float* __restrict__ bias,
                                                 float* __restrict__ outf,
                                                 __bf16* __restrict__ q,
                                                 __bf16* __restrict__ k,
                                                 __bf16* __restrict__ v,
                                                 int N) {
  const int K = 1024;
  __shared__ __align__(16) __bf16 Alds[128 * 40];   // [row][k] stride 40
  __shared__ __align__(16) __bf16 Blds[128 * 40];   // [n][k]   stride 40

  const int t    = threadIdx.x;
  const int lane = t & 31;
  const int wv   = t >> 5;
  const int bm   = blockIdx.y * 128;
  const int bn   = blockIdx.x * 128;
  const int wm   = (wv & 1) * 64;
  const int wn   = (wv >> 1) * 32;
  const int half = lane >> 4;
  const int ln16 = lane & 15;
  const int c0   = half * 8;          // ISA 16-bit A/B-frag K grouping

  v8f acc[4][2];
#pragma unroll
  for (int i = 0; i < 4; ++i)
#pragma unroll
    for (int j = 0; j < 2; ++j) acc[i][j] = v8f_zero();

  for (int kk = 0; kk < K; kk += 32) {
    __syncthreads();
#pragma unroll
    for (int it = 0; it < 2; ++it) {
      int u = t + it * 256;
      int row = u >> 2, uc = u & 3;
      async_cp_b128(Alds + row * 40 + uc * 8,
                    A + (size_t)(bm + row) * K + kk + uc * 8);
      async_cp_b128(Blds + row * 40 + uc * 8,
                    Bt + (size_t)(bn + row) * K + kk + uc * 8);
    }
    async_wait0();
    __syncthreads();

    v16bf af[4], bfr[2];
#pragma unroll
    for (int mt = 0; mt < 4; ++mt) {
      int rr = wm + mt * 16 + ln16;
      v8bf lo = *(const v8bf*)(Alds + rr * 40 + c0);
      v8bf hi = *(const v8bf*)(Alds + rr * 40 + c0 + 16);
      af[mt] = combine16(lo, hi);
    }
#pragma unroll
    for (int nt = 0; nt < 2; ++nt) {
      int cc = wn + nt * 16 + ln16;
      v8bf lo = *(const v8bf*)(Blds + cc * 40 + c0);
      v8bf hi = *(const v8bf*)(Blds + cc * 40 + c0 + 16);
      bfr[nt] = combine16(lo, hi);
    }
#pragma unroll
    for (int mt = 0; mt < 4; ++mt)
#pragma unroll
      for (int nt = 0; nt < 2; ++nt)
        acc[mt][nt] = WMMA_BF16(af[mt], bfr[nt], acc[mt][nt]);
  }

#pragma unroll
  for (int mt = 0; mt < 4; ++mt)
#pragma unroll
    for (int nt = 0; nt < 2; ++nt) {
#pragma unroll
      for (int i = 0; i < 8; ++i) {
        int row = bm + wm + mt * 16 + (half ? i + 8 : i);
        int col = bn + wn + nt * 16 + ln16;
        float val = acc[mt][nt][i] + bias[col];
        if (EPI == 0) {
          outf[(size_t)row * N + col] = val;
        } else {
          int which = col >> 10;            // 0:q 1:k 2:v
          int h = (col >> 6) & 15;
          int d = col & 63;
          int b = row >> 11, s = row & 2047;
          if (which == 0) {
            q[((size_t)((b * 16 + h) * 2048 + s)) * 64 + d] =
                (__bf16)(val * 0.125f);     // fold 1/sqrt(64)
          } else if (which == 1) {
            k[((size_t)((b * 16 + h) * 2048 + s)) * 64 + d] = (__bf16)val;
          } else {
            v[((size_t)((b * 16 + h) * 64 + d)) * 2048 + s] = (__bf16)val;
          }
        }
      }
    }
}

// ---------------------------------------------------------------------------
// Flash attention: grid (S/128, B*H), 8 waves; each wave owns 16 Q rows.
// K tile [32][64] -> Klds (stride 72) and V tile [64][32] -> Vt (stride 40)
// DMA'd by the Tensor Data Mover (TENSORcnt) with hardware padding; per-lane
// async copies as fallback. Row-sums of P computed with a ones-matrix WMMA.
// ---------------------------------------------------------------------------
__global__ void __launch_bounds__(256) flash_attn(const __bf16* __restrict__ Q,
                                                  const __bf16* __restrict__ Kb,
                                                  const __bf16* __restrict__ Vt_g,
                                                  __bf16* __restrict__ ctx) {
  const int S = 2048, Hd = 64;
  __shared__ __align__(16) __bf16 Klds[32 * 72];       // [kv][d]  stride 72
  __shared__ __align__(16) __bf16 Vt[64 * 40];         // [d][kv]  stride 40
  __shared__ __align__(16) __bf16 Plds[8 * 16 * 32];   // per-wave [16][32]

  const int t = threadIdx.x, lane = t & 31, wv = t >> 5;
  const int bh = blockIdx.y;       // b*16 + h
  const int b = bh >> 4, h = bh & 15;
  const int qb = blockIdx.x * 128 + wv * 16;
  const int half = lane >> 4;
  const int ln16 = lane & 15;
  const int c0 = half * 8;

  const __bf16* Qg = Q + (size_t)bh * S * Hd;
  const __bf16* Kg = Kb + (size_t)bh * S * Hd;
  const __bf16* Vg = Vt_g + (size_t)bh * Hd * S;    // [64][2048]

  // Q fragments [16 x 64] -> two 16x32 A-frags, straight from global
  v16bf qf[2];
  {
    int r = qb + ln16;
#pragma unroll
    for (int c = 0; c < 2; ++c) {
      int cc = c0 + c * 32;
      v8bf lo = *(const v8bf*)(Qg + (size_t)r * Hd + cc);
      v8bf hi = *(const v8bf*)(Qg + (size_t)r * Hd + cc + 16);
      qf[c] = combine16(lo, hi);
    }
  }

  v16bf ones16;
#pragma unroll
  for (int i = 0; i < 16; ++i) ones16[i] = (__bf16)1.0f;

  float m_run[8], l_run[8];
#pragma unroll
  for (int i = 0; i < 8; ++i) { m_run[i] = -3.0e38f; l_run[i] = 0.0f; }
  v8f o[4];
#pragma unroll
  for (int i = 0; i < 4; ++i) o[i] = v8f_zero();

  __bf16* Pw = Plds + wv * 16 * 32;

  for (int kt = 0; kt < S / 32; ++kt) {
    __syncthreads();
#if HAVE_TDM
    if (wv == 0) {
      // K tile: 32 rows x 64 elems; burst 32 dw (code 4), pad 4 dw (code 3) -> stride 72
      tdm_load_2d_bf16((unsigned)(unsigned long long)(const void*)Klds,
                       Kg + (size_t)kt * 32 * Hd,
                       /*t_d0=*/Hd, /*t_d1=*/S, /*d0_stride=*/Hd,
                       /*tile_d0=*/Hd, /*tile_d1=*/32,
                       /*pad_interval=*/4, /*pad_amount=*/3);
      // V tile: 64 rows x 32 elems; burst 16 dw (code 3), pad 4 dw (code 3) -> stride 40
      tdm_load_2d_bf16((unsigned)(unsigned long long)(const void*)Vt,
                       Vg + (size_t)kt * 32,
                       /*t_d0=*/S, /*t_d1=*/Hd, /*d0_stride=*/S,
                       /*tile_d0=*/32, /*tile_d1=*/Hd,
                       /*pad_interval=*/3, /*pad_amount=*/3);
      tensor_wait0();
    }
#else
    {
      int kr = t >> 3, kc = t & 7;
      async_cp_b128(Klds + kr * 72 + kc * 8,
                    Kg + (size_t)(kt * 32 + kr) * Hd + kc * 8);
      int d = t >> 2, uc = t & 3;
      async_cp_b128(Vt + d * 40 + uc * 8,
                    Vg + (size_t)d * S + kt * 32 + uc * 8);
    }
    async_wait0();
#endif
    __syncthreads();

    // ---- scores S = Q @ K^T : [16 x 32]
    v8f s0 = v8f_zero(), s1 = v8f_zero();
#pragma unroll
    for (int c = 0; c < 2; ++c) {
      int cc = c0 + c * 32;
      {
        v8bf lo = *(const v8bf*)(Klds + ln16 * 72 + cc);
        v8bf hi = *(const v8bf*)(Klds + ln16 * 72 + cc + 16);
        s0 = WMMA_BF16(qf[c], combine16(lo, hi), s0);
      }
      {
        v8bf lo = *(const v8bf*)(Klds + (16 + ln16) * 72 + cc);
        v8bf hi = *(const v8bf*)(Klds + (16 + ln16) * 72 + cc + 16);
        s1 = WMMA_BF16(qf[c], combine16(lo, hi), s1);
      }
    }

    // ---- row max (butterfly over 16 lanes), P = exp(s - m), stage P
    float fsc[8];
#pragma unroll
    for (int r = 0; r < 8; ++r) {
      float mx = fmaxf(s0[r], s1[r]);
#pragma unroll
      for (int off = 8; off >= 1; off >>= 1)
        mx = fmaxf(mx, __shfl_xor(mx, off, 16));
      float mnew = fmaxf(m_run[r], mx);
      fsc[r] = __expf(m_run[r] - mnew);
      m_run[r] = mnew;
      float p0 = __expf(s0[r] - mnew);
      float p1 = __expf(s1[r] - mnew);
      int prow = half ? (r + 8) : r;
      Pw[prow * 32 + ln16]      = (__bf16)p0;
      Pw[prow * 32 + 16 + ln16] = (__bf16)p1;
    }

    // ---- P A-fragment reload (wave-private LDS)
    v16bf pf;
    {
      v8bf lo = *(const v8bf*)(Pw + ln16 * 32 + c0);
      v8bf hi = *(const v8bf*)(Pw + ln16 * 32 + c0 + 16);
      pf = combine16(lo, hi);
    }

    // ---- row sums via matrix engine: P @ ones -> rowsum in C layout
    v8f rs = WMMA_BF16(pf, ones16, v8f_zero());
#pragma unroll
    for (int r = 0; r < 8; ++r) {
      l_run[r] = l_run[r] * fsc[r] + rs[r];
#pragma unroll
      for (int t2 = 0; t2 < 4; ++t2) o[t2][r] *= fsc[r];
    }

    // ---- O += P @ V
#pragma unroll
    for (int t2 = 0; t2 < 4; ++t2) {
      int n = t2 * 16 + ln16;
      v8bf lo = *(const v8bf*)(Vt + n * 40 + c0);
      v8bf hi = *(const v8bf*)(Vt + n * 40 + c0 + 16);
      o[t2] = WMMA_BF16(pf, combine16(lo, hi), o[t2]);
    }
  }

  // ---- normalize and store ctx [B, S, H*64] bf16
#pragma unroll
  for (int r = 0; r < 8; ++r) {
    float inv = 1.0f / l_run[r];
    int row = qb + (half ? r + 8 : r);
#pragma unroll
    for (int t2 = 0; t2 < 4; ++t2) {
      float val = o[t2][r] * inv;
      ctx[((size_t)(b * S + row)) * 1024 + h * 64 + t2 * 16 + ln16] = (__bf16)val;
    }
  }
}

// ---------------------------------------------------------------------------
extern "C" void kernel_launch(void* const* d_in, const int* in_sizes, int n_in,
                              void* d_out, int out_size, void* d_ws, size_t ws_size,
                              hipStream_t stream) {
  const float* x     = (const float*)d_in[0];  // [4,2048,1024]
  const float* w_qkv = (const float*)d_in[1];  // [1024,3072]
  const float* b_qkv = (const float*)d_in[2];  // [3072]
  const float* w_out = (const float*)d_in[3];  // [1024,1024]
  const float* b_out = (const float*)d_in[4];  // [1024]
  float* out = (float*)d_out;                  // [4,2048,1024]

  char* ws = (char*)d_ws;
  __bf16* xb   = (__bf16*)(ws);                // 16 MB  [8192][1024]
  __bf16* wqbT = (__bf16*)(ws + 16777216);     // 6 MB   [3072][1024]
  __bf16* wobT = (__bf16*)(ws + 23068672);     // 2 MB   [1024][1024]
  __bf16* q    = (__bf16*)(ws + 25165824);     // 16 MB  [4,16,2048,64]
  __bf16* k    = (__bf16*)(ws + 41943040);     // 16 MB  [4,16,2048,64]
  __bf16* v    = (__bf16*)(ws + 58720256);     // 16 MB  [4,16,64,2048] (transposed)
  __bf16* ctx  = (__bf16*)(ws + 75497472);     // 16 MB  [8192][1024]

  cvt_f32_bf16<<<8192, 256, 0, stream>>>(x, xb, 8388608 / 4);
  cvt_t_f32_bf16<<<12288, 256, 0, stream>>>(w_qkv, wqbT, 3072);
  cvt_t_f32_bf16<<<4096, 256, 0, stream>>>(w_out, wobT, 1024);

  dim3 g1(24, 64);  // N/128, M/128
  gemm_bf16<1><<<g1, 256, 0, stream>>>(xb, wqbT, b_qkv, nullptr, q, k, v, 3072);

  dim3 g2(16, 64);  // S/128, B*H
  flash_attn<<<g2, 256, 0, stream>>>(q, k, v, ctx);

  dim3 g3(8, 64);   // N/128, M/128
  gemm_bf16<0><<<g3, 256, 0, stream>>>(ctx, wobT, b_out, out, nullptr, nullptr, nullptr, 1024);
}
